// Net_36112085025445
// MI455X (gfx1250) — matrix operand order repaired and checked by reading, not compiled, not run on hardware
//
#include <hip/hip_runtime.h>
#include <math.h>

// ---------------- problem constants (match reference) ----------------
#define NNODES 65536
#define DIM    128
#define NEDGES 524288
#define NB     64
#define NPG0   1024
#define KK1    820
#define KK2    656
#define KK3    525
#define BN_EPS 1e-5f
#define NEG_INF (-3.402823466e38f)
#define LDIM   132   // LDS row stride (floats): 528B = 16B-aligned, 132%64==4 -> conflict-free banks

typedef float    v2f  __attribute__((ext_vector_type(2)));
typedef float    v8f  __attribute__((ext_vector_type(8)));
typedef _Float16 v16h __attribute__((ext_vector_type(16)));
typedef int      v4i  __attribute__((vector_size(4 * sizeof(int))));  // matches builtin param type

#define AS_GLOBAL __attribute__((address_space(1)))
#define AS_LDS    __attribute__((address_space(3)))

#if defined(__has_builtin)
#if __has_builtin(__builtin_amdgcn_wmma_f32_16x16x4_f32)
#define USE_F32_WMMA 1
#endif
#if __has_builtin(__builtin_amdgcn_global_load_async_to_lds_b128)
#define HAVE_ASYNC_LDS 1
#endif
#endif

__device__ __forceinline__ void wait_async0() {
#if defined(__has_builtin) && __has_builtin(__builtin_amdgcn_s_wait_asynccnt)
    __builtin_amdgcn_s_wait_asynccnt(0);
#else
    asm volatile("s_wait_asynccnt 0" ::: "memory");
#endif
}

// ---------------- generic fills ----------------
__global__ void fill_f32_kernel(float* p, float v, long n) {
    long i = (long)blockIdx.x * blockDim.x + threadIdx.x;
    long s = (long)gridDim.x * blockDim.x;
    for (; i < n; i += s) p[i] = v;
}
__global__ void fill_i32_kernel(int* p, int v, long n) {
    long i = (long)blockIdx.x * blockDim.x + threadIdx.x;
    long s = (long)gridDim.x * blockDim.x;
    for (; i < n; i += s) p[i] = v;
}

// ---------------- stage 0: edges + embedding gather + weight transpose ----------------
__global__ void init_edges_kernel(const int* __restrict__ ei,
                                  int* __restrict__ src, int* __restrict__ dst,
                                  float* __restrict__ em) {
    int e = blockIdx.x * blockDim.x + threadIdx.x;
    if (e >= NEDGES) return;
    src[e] = ei[e];
    dst[e] = ei[NEDGES + e];
    em[e]  = 1.0f;
}

__global__ void gather_embed_kernel(const int* __restrict__ node_idx,
                                    const float* __restrict__ emb,
                                    float* __restrict__ x, int M) {
    int tid = blockIdx.x * blockDim.x + threadIdx.x;
    int i = tid >> 5, q = tid & 31;          // 32 float4 per 128-f32 row
    if (i >= M) return;
    int v = node_idx[i];
    float4 t = ((const float4*)(emb + (size_t)v * DIM))[q];
    ((float4*)(x + (size_t)i * DIM))[q] = t;
}

// W is (out,in) row-major; WT[k*128 + o] = W[o*128 + k] so B-frag reads coalesce.
__global__ void transpose128_kernel(const float* __restrict__ W, float* __restrict__ WT) {
    int idx = blockIdx.x * blockDim.x + threadIdx.x;   // 16384 elements
    int o = idx >> 7, k = idx & 127;
    WT[(size_t)k * DIM + o] = W[(size_t)o * DIM + k];
}

// ---------------- segment mean ----------------
__global__ void scatter_add_kernel(const float* __restrict__ x,
                                   const int* __restrict__ src,
                                   const int* __restrict__ dst,
                                   const float* __restrict__ em,
                                   float* __restrict__ agg,
                                   float* __restrict__ cnt, int ne) {
    int tid = blockIdx.x * blockDim.x + threadIdx.x;
    int e = tid >> 5, q = tid & 31;
    if (e >= ne) return;
    float w = em[e];
    if (w == 0.0f) return;                   // zero contribution either way
    int s = src[e], d = dst[e];
    float4 v = ((const float4*)(x + (size_t)s * DIM))[q];
    float* a = agg + (size_t)d * DIM + q * 4;
    atomicAdd(a + 0, v.x * w);
    atomicAdd(a + 1, v.y * w);
    atomicAdd(a + 2, v.z * w);
    atomicAdd(a + 3, v.w * w);
    if (q == 0) atomicAdd(cnt + d, w);
}

__global__ void mean_norm_kernel(float* __restrict__ agg,
                                 const float* __restrict__ cnt, int M) {
    int tid = blockIdx.x * blockDim.x + threadIdx.x;
    int i = tid >> 5, q = tid & 31;
    if (i >= M) return;
    float inv = 1.0f / fmaxf(cnt[i], 1.0f);
    float4* p = (float4*)(agg + (size_t)i * DIM) + q;
    float4 v = *p;
    v.x *= inv; v.y *= inv; v.z *= inv; v.w *= inv;
    *p = v;
}

// ---------------- SAGE GEMM via WMMA (A tiles staged in LDS, B transposed) ----------------
// Y[m, n] = relu( bias[n] + sum_k mean[m,k]*WL[n,k] + sum_k X[m,k]*WR[n,k] )
// grid.x = M/16 (M always a multiple of 16 here); block = 256 = 8 waves;
// wave w owns the 16x16 output tile at columns [16w, 16w+16).
__global__ void sage_gemm_kernel(const float* __restrict__ A1,    // mean, M x 128
                                 const float* __restrict__ A2,    // x,    M x 128
                                 const float* __restrict__ WLT,   // 128 x 128 transposed [k][n]
                                 const float* __restrict__ WRT,
                                 const float* __restrict__ bias,
                                 float* __restrict__ Y, int M) {
    __shared__ float lA1[16 * LDIM];
    __shared__ float lA2[16 * LDIM];

    int t = threadIdx.x;
    int wave = t >> 5;
    int lane = t & 31;
    int m0 = blockIdx.x * 16;
    int n0 = wave * 16;
    int ncol = n0 + (lane & 15);
    int hi = lane >> 4;                       // 0 for lanes 0-15, 1 for 16-31

    // ---- cooperative stage of both A tiles: 2 x (16 rows x 128 f32) ----
    // 512 float4 per tile; 256 threads -> 2 each; fully coalesced 16B transfers.
#pragma unroll
    for (int idx = t; idx < 512; idx += 256) {
        int row = idx >> 5, q = idx & 31;
        const float4* g1 = (const float4*)(A1 + (size_t)(m0 + row) * DIM) + q;
        const float4* g2 = (const float4*)(A2 + (size_t)(m0 + row) * DIM) + q;
        float* d1 = lA1 + row * LDIM + q * 4;     // 528B row stride: 16B aligned
        float* d2 = lA2 + row * LDIM + q * 4;
#ifdef HAVE_ASYNC_LDS
        __builtin_amdgcn_global_load_async_to_lds_b128(
            (AS_GLOBAL v4i*)g1, (AS_LDS v4i*)d1, 0, 0);
        __builtin_amdgcn_global_load_async_to_lds_b128(
            (AS_GLOBAL v4i*)g2, (AS_LDS v4i*)d2, 0, 0);
#else
        *(float4*)d1 = *g1;
        *(float4*)d2 = *g2;
#endif
    }
#ifdef HAVE_ASYNC_LDS
    wait_async0();
#endif
    __syncthreads();

    v8f acc;
    float bv = bias[ncol];
#pragma unroll
    for (int r = 0; r < 8; ++r) acc[r] = bv;

    const float* la1 = lA1 + (lane & 15) * LDIM;
    const float* la2 = lA2 + (lane & 15) * LDIM;

#ifdef USE_F32_WMMA
    // f32 16x16x4: A frag lane l holds K = k+2*hi, k+2*hi+1 of row M=(l&15);
    // B frag lane l holds K = k+2*hi, k+2*hi+1 of column N=n0+(l&15).
    int kh = hi * 2;
#pragma unroll 4
    for (int k = 0; k < DIM; k += 4) {
        v2f a, b;
        a.x = la1[k + kh]; a.y = la1[k + kh + 1];
        b.x = WLT[(size_t)(k + kh) * DIM + ncol];
        b.y = WLT[(size_t)(k + kh + 1) * DIM + ncol];
        acc = __builtin_amdgcn_wmma_f32_16x16x4_f32(false, a, false, b,
                                                    (short)0, acc, false, false);
        a.x = la2[k + kh]; a.y = la2[k + kh + 1];
        b.x = WRT[(size_t)(k + kh) * DIM + ncol];
        b.y = WRT[(size_t)(k + kh + 1) * DIM + ncol];
        acc = __builtin_amdgcn_wmma_f32_16x16x4_f32(false, a, false, b,
                                                    (short)0, acc, false, false);
    }
#else
    // f16 16x16x32 fallback (f32 accumulate).
    int ga = hi * 8;
    int gb = hi * 16;
#pragma unroll
    for (int k0 = 0; k0 < DIM; k0 += 32) {
        v16h a, b;
#pragma unroll
        for (int j = 0; j < 8; ++j) {
            a[j]     = (_Float16)la1[k0 + gb + j];
            a[8 + j] = (_Float16)la1[k0 + gb + 8 + j];
        }
#pragma unroll
        for (int j = 0; j < 16; ++j)
            b[j] = (_Float16)WLT[(size_t)(k0 + gb + j) * DIM + ncol];
        acc = __builtin_amdgcn_wmma_f32_16x16x32_f16(false, a, false, b,
                                                     (short)0, acc, false, false);
#pragma unroll
        for (int j = 0; j < 8; ++j) {
            a[j]     = (_Float16)la2[k0 + gb + j];
            a[8 + j] = (_Float16)la2[k0 + gb + 8 + j];
        }
#pragma unroll
        for (int j = 0; j < 16; ++j)
            b[j] = (_Float16)WRT[(size_t)(k0 + gb + j) * DIM + ncol];
        acc = __builtin_amdgcn_wmma_f32_16x16x32_f16(false, a, false, b,
                                                     (short)0, acc, false, false);
    }
#endif

    // C/D layout: VGPR r, lanes 0-15 -> M=r, lanes 16-31 -> M=r+8; N=lane&15
#pragma unroll
    for (int r = 0; r < 8; ++r) {
        int m = m0 + r + 8 * hi;
        float v = acc[r];
        Y[(size_t)m * DIM + ncol] = v > 0.0f ? v : 0.0f;
    }
}

// ---------------- pooling: scores, top-k, gather ----------------
__global__ void pool_norm_kernel(const float* __restrict__ w, float* __restrict__ invn) {
    __shared__ float s[DIM];
    int t = threadIdx.x;
    float v = w[t];
    s[t] = v * v;
    __syncthreads();
    for (int off = DIM / 2; off > 0; off >>= 1) {
        if (t < off) s[t] += s[t + off];
        __syncthreads();
    }
    if (t == 0) *invn = rsqrtf(s[0]);
}

__global__ void score_kernel(const float* __restrict__ x,
                             const float* __restrict__ w,
                             const float* __restrict__ invn,
                             float* __restrict__ sc, int M) {
    int wave = threadIdx.x >> 5, lane = threadIdx.x & 31;
    int row = blockIdx.x * 8 + wave;
    if (row >= M) return;
    const float* xr = x + (size_t)row * DIM;
    float s = 0.0f;
#pragma unroll
    for (int c = lane; c < DIM; c += 32) s += xr[c] * w[c];
#pragma unroll
    for (int off = 16; off > 0; off >>= 1) s += __shfl_down(s, off, 32);
    if (lane == 0) sc[row] = s * (*invn);
}

// One block (1024 threads) per graph: bitonic sort (score,idx) descending in LDS.
__global__ __launch_bounds__(1024)
void topk_kernel(const float* __restrict__ sc, int npg, int k,
                 float* __restrict__ vals, int* __restrict__ perm) {
    __shared__ float sv[NPG0];
    __shared__ int   si[NPG0];
    int g = blockIdx.x, t = threadIdx.x;
    sv[t] = (t < npg) ? sc[g * npg + t] : NEG_INF;
    si[t] = t;
    __syncthreads();
    for (int size = 2; size <= NPG0; size <<= 1) {
        for (int stride = size >> 1; stride > 0; stride >>= 1) {
            int p = t ^ stride;
            if (p > t) {
                bool desc = ((t & size) == 0);
                float v1 = sv[t], v2 = sv[p];
                bool sw;
                if (v1 == v2) sw = desc ? (si[t] > si[p]) : (si[t] < si[p]);
                else          sw = desc ? (v1 < v2) : (v1 > v2);
                if (sw) {
                    sv[t] = v2; sv[p] = v1;
                    int ti = si[t]; si[t] = si[p]; si[p] = ti;
                }
            }
            __syncthreads();
        }
    }
    if (t < k) {
        vals[g * k + t] = sv[t];
        perm[g * k + t] = g * npg + si[t];
    }
}

__global__ void setid_kernel(const int* __restrict__ perm, int* __restrict__ newid, int n) {
    int j = blockIdx.x * blockDim.x + threadIdx.x;
    if (j < n) newid[perm[j]] = j;
}

__global__ void pool_apply_kernel(const float* __restrict__ y,
                                  const float* __restrict__ vals,
                                  const int* __restrict__ perm,
                                  float* __restrict__ xp, int rows) {
    int tid = blockIdx.x * blockDim.x + threadIdx.x;
    int j = tid >> 5, q = tid & 31;
    if (j >= rows) return;
    float s = tanhf(vals[j]);
    int p = perm[j];
    float4 v = ((const float4*)(y + (size_t)p * DIM))[q];
    v.x *= s; v.y *= s; v.z *= s; v.w *= s;
    ((float4*)(xp + (size_t)j * DIM))[q] = v;
}

__global__ void remap_kernel(const int* __restrict__ src, const int* __restrict__ dst,
                             const float* __restrict__ em, const int* __restrict__ newid,
                             int* __restrict__ src2, int* __restrict__ dst2,
                             float* __restrict__ em2, int ne) {
    int e = blockIdx.x * blockDim.x + threadIdx.x;
    if (e >= ne) return;
    int ns = newid[src[e]], nd = newid[dst[e]];
    float w = em[e];
    if (ns < 0 || nd < 0) w = 0.0f;
    src2[e] = ns < 0 ? 0 : ns;
    dst2[e] = nd < 0 ? 0 : nd;
    em2[e] = w;
}

// ---------------- readout + head ----------------
__global__ void readout_kernel(const float* __restrict__ xp, float* __restrict__ r, int k) {
    int g = blockIdx.x, c = threadIdx.x;  // 128 threads
    const float* base = xp + (size_t)g * k * DIM + c;
    float mx = NEG_INF, sum = 0.0f;
    for (int i = 0; i < k; ++i) {
        float v = base[(size_t)i * DIM];
        mx = fmaxf(mx, v);
        sum += v;
    }
    r[g * 256 + c] = mx;
    r[g * 256 + 128 + c] = sum / (float)k;
}

__global__ void sum3_kernel(const float* a, const float* b, const float* c,
                            float* h, int n) {
    int i = blockIdx.x * blockDim.x + threadIdx.x;
    if (i < n) h[i] = a[i] + b[i] + c[i];
}

__global__ void fc_kernel(const float* __restrict__ in, const float* __restrict__ W,
                          const float* __restrict__ bias, float* __restrict__ out,
                          int rows, int K, int O) {
    int idx = blockIdx.x * blockDim.x + threadIdx.x;
    if (idx >= rows * O) return;
    int i = idx / O, o = idx % O;
    float s = bias[o];
    const float* ir = in + (size_t)i * K;
    const float* wr = W + (size_t)o * K;
    for (int kk = 0; kk < K; ++kk) s += ir[kk] * wr[kk];
    out[idx] = s;
}

__global__ void bn_relu_kernel(float* __restrict__ x, const float* __restrict__ g,
                               const float* __restrict__ b, int rows, int O) {
    int o = blockIdx.x * blockDim.x + threadIdx.x;
    if (o >= O) return;
    float m = 0.0f;
    for (int i = 0; i < rows; ++i) m += x[(size_t)i * O + o];
    m /= rows;
    float v = 0.0f;
    for (int i = 0; i < rows; ++i) {
        float d = x[(size_t)i * O + o] - m;
        v += d * d;
    }
    v /= rows;
    float inv = rsqrtf(v + BN_EPS) * g[o];
    for (int i = 0; i < rows; ++i) {
        float y = (x[(size_t)i * O + o] - m) * inv + b[o];
        x[(size_t)i * O + o] = y > 0.0f ? y : 0.0f;
    }
}

__global__ void head_kernel(const float* __restrict__ h, const float* __restrict__ lw,
                            const float* __restrict__ lb, float* __restrict__ out,
                            int rows, int K) {
    int i = blockIdx.x * blockDim.x + threadIdx.x;
    if (i >= rows) return;
    float s = lb[0];
    for (int k = 0; k < K; ++k) s += h[(size_t)i * K + k] * lw[k];
    out[i] = 1.0f / (1.0f + expf(-s));
}

// ---------------- host orchestration ----------------
static inline int cdiv_i(long a, int b) { return (int)((a + b - 1) / b); }

static void run_sage(const float* Xin, float* Yout, float* AGG, float* CNT,
                     const int* SRC, const int* DST, const float* EM,
                     const float* WLT, const float* BL, const float* WRT,
                     int M, hipStream_t st) {
    fill_f32_kernel<<<cdiv_i((long)M * DIM, 256), 256, 0, st>>>(AGG, 0.0f, (long)M * DIM);
    fill_f32_kernel<<<cdiv_i(M, 256), 256, 0, st>>>(CNT, 0.0f, (long)M);
    scatter_add_kernel<<<cdiv_i((long)NEDGES * 32, 256), 256, 0, st>>>(Xin, SRC, DST, EM, AGG, CNT, NEDGES);
    mean_norm_kernel<<<cdiv_i((long)M * 32, 256), 256, 0, st>>>(AGG, CNT, M);
    sage_gemm_kernel<<<M / 16, 256, 0, st>>>(AGG, Xin, WLT, WRT, BL, Yout, M);
}

extern "C" void kernel_launch(void* const* d_in, const int* in_sizes, int n_in,
                              void* d_out, int out_size, void* d_ws, size_t ws_size,
                              hipStream_t stream) {
    const int*   node_idx = (const int*)d_in[0];
    const int*   edge_idx = (const int*)d_in[1];
    const float* emb      = (const float*)d_in[2];
    const float* w1l = (const float*)d_in[3];
    const float* b1l = (const float*)d_in[4];
    const float* w1r = (const float*)d_in[5];
    const float* pw  = (const float*)d_in[6];
    const float* w2l = (const float*)d_in[7];
    const float* b2l = (const float*)d_in[8];
    const float* w2r = (const float*)d_in[9];
    const float* f1w = (const float*)d_in[10];
    const float* f1b = (const float*)d_in[11];
    const float* g1  = (const float*)d_in[12];
    const float* be1 = (const float*)d_in[13];
    const float* f2w = (const float*)d_in[14];
    const float* f2b = (const float*)d_in[15];
    const float* g2  = (const float*)d_in[16];
    const float* be2 = (const float*)d_in[17];
    const float* lw  = (const float*)d_in[18];
    const float* lb  = (const float*)d_in[19];
    float* out = (float*)d_out;

    // ---- workspace layout (bytes) ----
    char* ws = (char*)d_ws;
    size_t off = 0;
    const size_t ROWBYTES = (size_t)NNODES * DIM * sizeof(float);
    float* X    = (float*)(ws + off); off += ROWBYTES;
    float* Y    = (float*)(ws + off); off += ROWBYTES;
    float* AGG  = (float*)(ws + off); off += ROWBYTES;
    float* CNT  = (float*)(ws + off); off += (size_t)NNODES * sizeof(float);
    float* SC   = (float*)(ws + off); off += (size_t)NNODES * sizeof(float);
    float* VALS = (float*)(ws + off); off += (size_t)NB * NPG0 * sizeof(float);
    int*   PERM = (int*)(ws + off);   off += (size_t)NB * NPG0 * sizeof(int);
    int*   NEWID= (int*)(ws + off);   off += (size_t)NNODES * sizeof(int);
    int*   SRCa = (int*)(ws + off);   off += (size_t)NEDGES * sizeof(int);
    int*   DSTa = (int*)(ws + off);   off += (size_t)NEDGES * sizeof(int);
    float* EMa  = (float*)(ws + off); off += (size_t)NEDGES * sizeof(float);
    int*   SRCb = (int*)(ws + off);   off += (size_t)NEDGES * sizeof(int);
    int*   DSTb = (int*)(ws + off);   off += (size_t)NEDGES * sizeof(int);
    float* EMb  = (float*)(ws + off); off += (size_t)NEDGES * sizeof(float);
    float* R1   = (float*)(ws + off); off += (size_t)NB * 256 * sizeof(float);
    float* R2   = (float*)(ws + off); off += (size_t)NB * 256 * sizeof(float);
    float* R3   = (float*)(ws + off); off += (size_t)NB * 256 * sizeof(float);
    float* H    = (float*)(ws + off); off += (size_t)NB * 256 * sizeof(float);
    float* H1   = (float*)(ws + off); off += (size_t)NB * 128 * sizeof(float);
    float* H2   = (float*)(ws + off); off += (size_t)NB * 64 * sizeof(float);
    float* W1LT = (float*)(ws + off); off += (size_t)DIM * DIM * sizeof(float);
    float* W1RT = (float*)(ws + off); off += (size_t)DIM * DIM * sizeof(float);
    float* W2LT = (float*)(ws + off); off += (size_t)DIM * DIM * sizeof(float);
    float* W2RT = (float*)(ws + off); off += (size_t)DIM * DIM * sizeof(float);
    float* INVN = (float*)(ws + off); off += sizeof(float);
    (void)ws_size; (void)in_sizes; (void)n_in; (void)out_size;

    // ---- stage 0 ----
    init_edges_kernel<<<cdiv_i(NEDGES, 256), 256, 0, stream>>>(edge_idx, SRCa, DSTa, EMa);
    pool_norm_kernel<<<1, DIM, 0, stream>>>(pw, INVN);
    transpose128_kernel<<<64, 256, 0, stream>>>(w1l, W1LT);
    transpose128_kernel<<<64, 256, 0, stream>>>(w1r, W1RT);
    transpose128_kernel<<<64, 256, 0, stream>>>(w2l, W2LT);
    transpose128_kernel<<<64, 256, 0, stream>>>(w2r, W2RT);
    gather_embed_kernel<<<cdiv_i((long)NNODES * 32, 256), 256, 0, stream>>>(node_idx, emb, X, NNODES);

    // ---- layer 1: SAGE on full graph, pool to K1 ----
    int M1 = NNODES;
    run_sage(X, Y, AGG, CNT, SRCa, DSTa, EMa, W1LT, b1l, W1RT, M1, stream);
    score_kernel<<<cdiv_i(M1, 8), 256, 0, stream>>>(Y, pw, INVN, SC, M1);
    topk_kernel<<<NB, NPG0, 0, stream>>>(SC, NPG0, KK1, VALS, PERM);
    fill_i32_kernel<<<cdiv_i(M1, 256), 256, 0, stream>>>(NEWID, -1, (long)M1);
    setid_kernel<<<cdiv_i(NB * KK1, 256), 256, 0, stream>>>(PERM, NEWID, NB * KK1);
    pool_apply_kernel<<<cdiv_i((long)NB * KK1 * 32, 256), 256, 0, stream>>>(Y, VALS, PERM, X, NB * KK1);
    remap_kernel<<<cdiv_i(NEDGES, 256), 256, 0, stream>>>(SRCa, DSTa, EMa, NEWID, SRCb, DSTb, EMb, NEDGES);
    readout_kernel<<<NB, DIM, 0, stream>>>(X, R1, KK1);

    // ---- layer 2: SAGE on pooled graph (M2 rows), pool to K2 ----
    int M2 = NB * KK1;
    run_sage(X, Y, AGG, CNT, SRCb, DSTb, EMb, W2LT, b2l, W2RT, M2, stream);
    score_kernel<<<cdiv_i(M2, 8), 256, 0, stream>>>(Y, pw, INVN, SC, M2);
    topk_kernel<<<NB, NPG0, 0, stream>>>(SC, KK1, KK2, VALS, PERM);
    fill_i32_kernel<<<cdiv_i(M2, 256), 256, 0, stream>>>(NEWID, -1, (long)M2);
    setid_kernel<<<cdiv_i(NB * KK2, 256), 256, 0, stream>>>(PERM, NEWID, NB * KK2);
    pool_apply_kernel<<<cdiv_i((long)NB * KK2 * 32, 256), 256, 0, stream>>>(Y, VALS, PERM, X, NB * KK2);
    remap_kernel<<<cdiv_i(NEDGES, 256), 256, 0, stream>>>(SRCb, DSTb, EMb, NEWID, SRCa, DSTa, EMa, NEDGES);
    readout_kernel<<<NB, DIM, 0, stream>>>(X, R2, KK2);

    // ---- layer 3: SAGE, pool to K3 (no edge remap needed after) ----
    int M3 = NB * KK2;
    run_sage(X, Y, AGG, CNT, SRCa, DSTa, EMa, W2LT, b2l, W2RT, M3, stream);
    score_kernel<<<cdiv_i(M3, 8), 256, 0, stream>>>(Y, pw, INVN, SC, M3);
    topk_kernel<<<NB, NPG0, 0, stream>>>(SC, KK2, KK3, VALS, PERM);
    pool_apply_kernel<<<cdiv_i((long)NB * KK3 * 32, 256), 256, 0, stream>>>(Y, VALS, PERM, X, NB * KK3);
    readout_kernel<<<NB, DIM, 0, stream>>>(X, R3, KK3);

    // ---- head ----
    sum3_kernel<<<cdiv_i(NB * 256, 256), 256, 0, stream>>>(R1, R2, R3, H, NB * 256);
    fc_kernel<<<cdiv_i(NB * 128, 256), 256, 0, stream>>>(H, f1w, f1b, H1, NB, 256, 128);
    bn_relu_kernel<<<1, 128, 0, stream>>>(H1, g1, be1, NB, 128);
    fc_kernel<<<cdiv_i(NB * 64, 256), 256, 0, stream>>>(H1, f2w, f2b, H2, NB, 128, 64);
    bn_relu_kernel<<<1, 64, 0, stream>>>(H2, g2, be2, NB, 64);
    head_kernel<<<1, 64, 0, stream>>>(H2, lw, lb, out, NB, 64);
}
